// SpectrogramLearner_32787780337889
// MI455X (gfx1250) — matrix operand order repaired and checked by reading, not compiled, not run on hardware
//
#include <hip/hip_runtime.h>

// ---------------------------------------------------------------------------
// Types for CDNA5 WMMA (wave32): V_WMMA_F32_16X16X32_F16
// ---------------------------------------------------------------------------
typedef _Float16 v8h  __attribute__((ext_vector_type(8)));
typedef _Float16 v16h __attribute__((ext_vector_type(16)));
typedef float    v8f  __attribute__((ext_vector_type(8)));

#define BATCH    64
#define HSZ      512
#define G4       2048      // 4 * HSZ (i,f,g,o)
#define TSTEPS   512
#define FB       513       // freq bins
#define KX1      544       // 513 padded up to 17*32 for K tiling
#define OUTN     513
#define OUTPAD   528       // 33 * 16

__device__ __forceinline__ float sigmoidf(float x) {
  return 1.0f / (1.0f + __expf(-x));
}

// Build the 16-half A fragment for 16-bit A 16x32 (contiguous K order, matches B):
// lanes 0-15: halves 0..15 = K[kk..kk+15]
// lanes16-31: halves 0..15 = K[kk+16..kk+31]
__device__ __forceinline__ v16h make_afrag(const _Float16* arow, int kk, int hsel) {
  v8h lo = *(const v8h*)(arow + kk + 2 * hsel);
  v8h hi = *(const v8h*)(arow + kk + 2 * hsel + 8);
  return __builtin_shufflevector(lo, hi, 0,1,2,3,4,5,6,7,8,9,10,11,12,13,14,15);
}

// Accumulate 4 gate tiles: acc[g] += A[16xK] * W[g*HSZ+ncol][K]  (B = W^T slice)
__device__ __forceinline__ void gemm4(const _Float16* __restrict__ A,
                                      const _Float16* __restrict__ W, int Kpad,
                                      int rowA, int ncol, v8f acc[4], int lane) {
  const int hsel = (lane & 16) ? 8 : 0;    // A K-sub-select per lane half
  const int bsel = (lane & 16) ? 16 : 0;   // B: lanes 16-31 take K+16..K+31
  const _Float16* arow = A + (size_t)rowA * Kpad;
  const _Float16* b0 = W + (size_t)(0 * HSZ + ncol) * Kpad + bsel;
  const _Float16* b1 = W + (size_t)(1 * HSZ + ncol) * Kpad + bsel;
  const _Float16* b2 = W + (size_t)(2 * HSZ + ncol) * Kpad + bsel;
  const _Float16* b3 = W + (size_t)(3 * HSZ + ncol) * Kpad + bsel;
  for (int kk = 0; kk < Kpad; kk += 32) {
    v16h a = make_afrag(arow, kk, hsel);
    acc[0] = __builtin_amdgcn_wmma_f32_16x16x32_f16(false, a, false,
              *(const v16h*)(b0 + kk), (short)0, acc[0], false, false);
    acc[1] = __builtin_amdgcn_wmma_f32_16x16x32_f16(false, a, false,
              *(const v16h*)(b1 + kk), (short)0, acc[1], false, false);
    acc[2] = __builtin_amdgcn_wmma_f32_16x16x32_f16(false, a, false,
              *(const v16h*)(b2 + kk), (short)0, acc[2], false, false);
    acc[3] = __builtin_amdgcn_wmma_f32_16x16x32_f16(false, a, false,
              *(const v16h*)(b3 + kk), (short)0, acc[3], false, false);
  }
}

// Single-tile GEMM for the FC layers: acc += A[16xK] * W[rowB][K]
__device__ __forceinline__ v8f gemm1(const _Float16* __restrict__ A,
                                     const _Float16* __restrict__ W, int Kpad,
                                     int rowA, int rowB, v8f acc, int lane) {
  const int hsel = (lane & 16) ? 8 : 0;
  const int bsel = (lane & 16) ? 16 : 0;
  const _Float16* arow = A + (size_t)rowA * Kpad;
  const _Float16* brow = W + (size_t)rowB * Kpad + bsel;
  for (int kk = 0; kk < Kpad; kk += 32) {
    v16h a = make_afrag(arow, kk, hsel);
    acc = __builtin_amdgcn_wmma_f32_16x16x32_f16(false, a, false,
            *(const v16h*)(brow + kk), (short)0, acc, false, false);
  }
  return acc;
}

// Fused LSTM cell nonlinearity on the 4 gate tiles held in registers.
// C/D layout: value v of lane l is (row = v + (l>=16 ? 8:0), col = l&15).
__device__ __forceinline__ void cell_update(v8f acc[4],
                                            const float* __restrict__ bih,
                                            const float* __restrict__ bhh,
                                            float* __restrict__ C,
                                            _Float16* __restrict__ Hout,
                                            int mbase, int ncol, int lane) {
  const float bi = bih[ncol]            + bhh[ncol];
  const float bf = bih[HSZ + ncol]      + bhh[HSZ + ncol];
  const float bg = bih[2 * HSZ + ncol]  + bhh[2 * HSZ + ncol];
  const float bo = bih[3 * HSZ + ncol]  + bhh[3 * HSZ + ncol];
  const int roff = (lane & 16) ? 8 : 0;
#pragma unroll
  for (int v = 0; v < 8; ++v) {
    const int idx = (mbase + roff + v) * HSZ + ncol;
    const float i = sigmoidf(acc[0][v] + bi);
    const float f = sigmoidf(acc[1][v] + bf);
    const float g = tanhf(acc[2][v] + bg);
    const float o = sigmoidf(acc[3][v] + bo);
    const float c = f * C[idx] + i * g;
    C[idx] = c;
    Hout[idx] = (_Float16)(o * tanhf(c));
  }
}

// Global (grid-wide) sense barrier via device-scope atomics.
__device__ __forceinline__ void grid_barrier(unsigned* bar) {
  __syncthreads();
  if (threadIdx.x == 0) {
    unsigned* cnt = bar;
    unsigned* gen = bar + 1;
    unsigned g = __atomic_load_n(gen, __ATOMIC_ACQUIRE);
    __threadfence();
    unsigned prev = __atomic_fetch_add(cnt, 1u, __ATOMIC_ACQ_REL);
    if (prev + 1u == gridDim.x) {
      __atomic_store_n(cnt, 0u, __ATOMIC_RELAXED);
      __atomic_fetch_add(gen, 1u, __ATOMIC_ACQ_REL);
    } else {
      while (__atomic_load_n(gen, __ATOMIC_ACQUIRE) == g) {
        __builtin_amdgcn_s_sleep(1);
      }
    }
  }
  __syncthreads();
}

// ---------------------------------------------------------------------------
// Prep kernels: f32 -> f16 with row/col zero-padding; x transpose; state init
// ---------------------------------------------------------------------------
__global__ void k_cvt_pad(const float* __restrict__ src, _Float16* __restrict__ dst,
                          int drows, int dcols, int srows, int scols) {
  int idx = blockIdx.x * blockDim.x + threadIdx.x;
  if (idx >= drows * dcols) return;
  int r = idx / dcols, c = idx - r * dcols;
  float v = (r < srows && c < scols) ? src[(size_t)r * scols + c] : 0.0f;
  dst[idx] = (_Float16)v;
}

// x[B][F][T] -> xf[T][B][KX1] (f16, zero-padded freq dim)
__global__ void k_cvt_x(const float* __restrict__ x, _Float16* __restrict__ xf) {
  int idx = blockIdx.x * blockDim.x + threadIdx.x;
  if (idx >= TSTEPS * BATCH * KX1) return;
  int t = idx / (BATCH * KX1);
  int rem = idx - t * (BATCH * KX1);
  int b = rem / KX1;
  int f = rem - b * KX1;
  float v = (f < FB) ? x[((size_t)b * FB + f) * TSTEPS + t] : 0.0f;
  xf[idx] = (_Float16)v;
}

__global__ void k_init(const float* h1, const float* c1, const float* h2,
                       const float* c2, const float* h3, const float* c3,
                       _Float16* H1, _Float16* H2, _Float16* H3,
                       float* C1, float* C2, float* C3, unsigned* bar) {
  int idx = blockIdx.x * blockDim.x + threadIdx.x;
  if (idx == 0) { bar[0] = 0u; bar[1] = 0u; }
  if (idx >= BATCH * HSZ) return;
  H1[idx] = (_Float16)h1[idx]; C1[idx] = c1[idx];
  H2[idx] = (_Float16)h2[idx]; C2[idx] = c2[idx];
  H3[idx] = (_Float16)h3[idx]; C3[idx] = c3[idx];
}

// ---------------------------------------------------------------------------
// Persistent recurrence kernel: 16 blocks x 256 threads = 128 wave32s.
// Each wave owns (mtile 0..3, ntile 0..31) -> 4 fused gate tiles per layer.
// h buffers are double-buffered (read t&1, write the other) so no barrier is
// needed between a layer's GEMM and its own cell update.
// ---------------------------------------------------------------------------
__global__ void __launch_bounds__(256, 1)
k_lstm(const _Float16* __restrict__ Xf,
       const _Float16* __restrict__ W1x, const _Float16* __restrict__ W1h,
       const _Float16* __restrict__ W2x, const _Float16* __restrict__ W2h,
       const _Float16* __restrict__ W3x, const _Float16* __restrict__ W3h,
       const float* __restrict__ bih1, const float* __restrict__ bhh1,
       const float* __restrict__ bih2, const float* __restrict__ bhh2,
       const float* __restrict__ bih3, const float* __restrict__ bhh3,
       _Float16* __restrict__ H1, _Float16* __restrict__ H2,
       _Float16* __restrict__ H3,
       float* __restrict__ C1, float* __restrict__ C2, float* __restrict__ C3,
       const _Float16* __restrict__ F1, const float* __restrict__ fc1_b,
       const _Float16* __restrict__ F2, const float* __restrict__ fc2_b,
       _Float16* __restrict__ HID, float* __restrict__ OUT, unsigned* bar) {
  const int lane  = threadIdx.x & 31;
  const int wid   = blockIdx.x * (blockDim.x >> 5) + (threadIdx.x >> 5); // 0..127
  const int mtile = wid >> 5;                    // 0..3  (batch tile)
  const int ntile = wid & 31;                    // 0..31 (hidden-col tile)
  const int mbase = mtile * 16;
  const int ncol  = ntile * 16 + (lane & 15);    // hidden column [0,512)
  const int rowA  = mbase + (lane & 15);         // batch row
  const int HB    = BATCH * HSZ;                 // 32768 elements per h buffer

#pragma unroll 1
  for (int t = 0; t < TSTEPS; ++t) {
    const int rd = t & 1, wr = rd ^ 1;
    { // layer 1: gates = x_t @ Wih1^T + h1 @ Whh1^T
      v8f acc[4] = {};
      gemm4(Xf + (size_t)t * BATCH * KX1, W1x, KX1, rowA, ncol, acc, lane);
      gemm4(H1 + rd * HB, W1h, HSZ, rowA, ncol, acc, lane);
      cell_update(acc, bih1, bhh1, C1, H1 + wr * HB, mbase, ncol, lane);
    }
    grid_barrier(bar);
    { // layer 2: input is the freshly-written h1
      v8f acc[4] = {};
      gemm4(H1 + wr * HB, W2x, HSZ, rowA, ncol, acc, lane);
      gemm4(H2 + rd * HB, W2h, HSZ, rowA, ncol, acc, lane);
      cell_update(acc, bih2, bhh2, C2, H2 + wr * HB, mbase, ncol, lane);
    }
    grid_barrier(bar);
    { // layer 3
      v8f acc[4] = {};
      gemm4(H2 + wr * HB, W3x, HSZ, rowA, ncol, acc, lane);
      gemm4(H3 + rd * HB, W3h, HSZ, rowA, ncol, acc, lane);
      cell_update(acc, bih3, bhh3, C3, H3 + wr * HB, mbase, ncol, lane);
    }
    grid_barrier(bar);
  }

  // Final h3 lives in buffer index (TSTEPS & 1) == 0.
  const _Float16* h3f = H3 + (TSTEPS & 1) * HB;

  { // FC1 + ReLU -> HID (f16), 128 tiles == 128 waves
    v8f acc = {};
    acc = gemm1(h3f, F1, HSZ, rowA, ncol, acc, lane);
    const float b = fc1_b[ncol];
    const int roff = (lane & 16) ? 8 : 0;
#pragma unroll
    for (int v = 0; v < 8; ++v) {
      float h = acc[v] + b;
      HID[(mbase + roff + v) * HSZ + ncol] = (_Float16)(h > 0.0f ? h : 0.0f);
    }
  }
  grid_barrier(bar);

  // FC2: out [64 x 513], padded to 4 x 33 = 132 tile jobs over 128 waves
  for (int j = wid; j < 4 * (OUTPAD / 16); j += 128) {
    const int mt = j / (OUTPAD / 16);
    const int nt = j - mt * (OUTPAD / 16);
    const int mb = mt * 16;
    const int nc = nt * 16 + (lane & 15);
    const int ra = mb + (lane & 15);
    v8f acc = {};
    acc = gemm1(HID, F2, HSZ, ra, nc, acc, lane);
    if (nc < OUTN) {
      const float b = fc2_b[nc];
      const int roff = (lane & 16) ? 8 : 0;
#pragma unroll
      for (int v = 0; v < 8; ++v)
        OUT[(mb + roff + v) * OUTN + nc] = acc[v] + b;
    }
  }
}

// ---------------------------------------------------------------------------
// Host launcher
// ---------------------------------------------------------------------------
extern "C" void kernel_launch(void* const* d_in, const int* in_sizes, int n_in,
                              void* d_out, int out_size, void* d_ws, size_t ws_size,
                              hipStream_t stream) {
  (void)in_sizes; (void)n_in; (void)out_size; (void)ws_size;
  const float* x    = (const float*)d_in[0];
  const float* Wih1 = (const float*)d_in[1];
  const float* Whh1 = (const float*)d_in[2];
  const float* bih1 = (const float*)d_in[3];
  const float* bhh1 = (const float*)d_in[4];
  const float* Wih2 = (const float*)d_in[5];
  const float* Whh2 = (const float*)d_in[6];
  const float* bih2 = (const float*)d_in[7];
  const float* bhh2 = (const float*)d_in[8];
  const float* Wih3 = (const float*)d_in[9];
  const float* Whh3 = (const float*)d_in[10];
  const float* bih3 = (const float*)d_in[11];
  const float* bhh3 = (const float*)d_in[12];
  const float* fc1w = (const float*)d_in[13];
  const float* fc1b = (const float*)d_in[14];
  const float* fc2w = (const float*)d_in[15];
  const float* fc2b = (const float*)d_in[16];
  const float* h1   = (const float*)d_in[17];
  const float* c1   = (const float*)d_in[18];
  const float* h2   = (const float*)d_in[19];
  const float* c2   = (const float*)d_in[20];
  const float* h3   = (const float*)d_in[21];
  const float* c3   = (const float*)d_in[22];
  float* out = (float*)d_out;

  // Carve the workspace (256-byte aligned slices).
  char* p = (char*)d_ws;
  auto carve = [&](size_t bytes) -> char* {
    char* r = p;
    p += (bytes + 255) & ~(size_t)255;
    return r;
  };
  _Float16* W1x = (_Float16*)carve((size_t)G4 * KX1 * 2);
  _Float16* W1h = (_Float16*)carve((size_t)G4 * HSZ * 2);
  _Float16* W2x = (_Float16*)carve((size_t)G4 * HSZ * 2);
  _Float16* W2h = (_Float16*)carve((size_t)G4 * HSZ * 2);
  _Float16* W3x = (_Float16*)carve((size_t)G4 * HSZ * 2);
  _Float16* W3h = (_Float16*)carve((size_t)G4 * HSZ * 2);
  _Float16* F1  = (_Float16*)carve((size_t)HSZ * HSZ * 2);
  _Float16* F2  = (_Float16*)carve((size_t)OUTPAD * HSZ * 2);
  _Float16* Xf  = (_Float16*)carve((size_t)TSTEPS * BATCH * KX1 * 2);
  _Float16* H1  = (_Float16*)carve((size_t)2 * BATCH * HSZ * 2);
  _Float16* H2  = (_Float16*)carve((size_t)2 * BATCH * HSZ * 2);
  _Float16* H3  = (_Float16*)carve((size_t)2 * BATCH * HSZ * 2);
  float*    C1  = (float*)carve((size_t)BATCH * HSZ * 4);
  float*    C2  = (float*)carve((size_t)BATCH * HSZ * 4);
  float*    C3  = (float*)carve((size_t)BATCH * HSZ * 4);
  _Float16* HID = (_Float16*)carve((size_t)BATCH * HSZ * 2);
  unsigned* BAR = (unsigned*)carve(256);

  const int TPB = 256;
  auto blocks = [&](long long n) { return (int)((n + TPB - 1) / TPB); };

  // Weight conversion / padding to f16
  k_cvt_pad<<<blocks((long long)G4 * KX1), TPB, 0, stream>>>(Wih1, W1x, G4, KX1, G4, FB);
  k_cvt_pad<<<blocks((long long)G4 * HSZ), TPB, 0, stream>>>(Whh1, W1h, G4, HSZ, G4, HSZ);
  k_cvt_pad<<<blocks((long long)G4 * HSZ), TPB, 0, stream>>>(Wih2, W2x, G4, HSZ, G4, HSZ);
  k_cvt_pad<<<blocks((long long)G4 * HSZ), TPB, 0, stream>>>(Whh2, W2h, G4, HSZ, G4, HSZ);
  k_cvt_pad<<<blocks((long long)G4 * HSZ), TPB, 0, stream>>>(Wih3, W3x, G4, HSZ, G4, HSZ);
  k_cvt_pad<<<blocks((long long)G4 * HSZ), TPB, 0, stream>>>(Whh3, W3h, G4, HSZ, G4, HSZ);
  k_cvt_pad<<<blocks((long long)HSZ * HSZ), TPB, 0, stream>>>(fc1w, F1, HSZ, HSZ, HSZ, HSZ);
  k_cvt_pad<<<blocks((long long)OUTPAD * HSZ), TPB, 0, stream>>>(fc2w, F2, OUTPAD, HSZ, OUTN, HSZ);

  // x transpose + f16 convert
  k_cvt_x<<<blocks((long long)TSTEPS * BATCH * KX1), TPB, 0, stream>>>(x, Xf);

  // State + barrier init (every call, so replays are deterministic)
  k_init<<<blocks((long long)BATCH * HSZ), TPB, 0, stream>>>(h1, c1, h2, c2, h3, c3,
                                                             H1, H2, H3, C1, C2, C3, BAR);

  // Persistent LSTM + FC kernel: 16 blocks x 256 threads = 128 waves
  k_lstm<<<16, 256, 0, stream>>>(Xf, W1x, W1h, W2x, W2h, W3x, W3h,
                                 bih1, bhh1, bih2, bhh2, bih3, bhh3,
                                 H1, H2, H3, C1, C2, C3,
                                 F1, fc1b, F2, fc2b, HID, out, BAR);
}
